// LabelAttention_88201448391027
// MI455X (gfx1250) — matrix operand order repaired and checked by reading, not compiled, not run on hardware
//
#include <hip/hip_runtime.h>
#include <hip/hip_bf16.h>
#include <math.h>

#define BS_  2
#define SEQ_ 4096
#define H_   768
#define L_   8921

typedef __attribute__((ext_vector_type(8)))  __bf16 v8bf;
typedef __attribute__((ext_vector_type(16))) __bf16 v16bf;
typedef __attribute__((ext_vector_type(8)))  float  v8f;

// ---- WMMA fragment helpers (CDNA5 wave32 layouts, CDNA5 ISA) ----
// A 16x32 bf16: lane&15 = row M; lane>>4 selects K-halves {0..15} vs {16..31}
__device__ __forceinline__ v16bf frag_a(const __bf16* rowp, int sel) {
  v8bf lo = *(const v8bf*)(rowp + sel * 16);       // K = sel*16 .. sel*16+7
  v8bf hi = *(const v8bf*)(rowp + sel * 16 + 8);   // K = sel*16+8 .. +7
  return __builtin_shufflevector(lo, hi, 0,1,2,3,4,5,6,7,8,9,10,11,12,13,14,15);
}
// B 32x16 bf16: lane&15 = col N; lanes>=16 hold K=16..31 (column-major LDS tile)
__device__ __forceinline__ v16bf frag_b(const __bf16* colp, int sel) {
  return *(const v16bf*)(colp + sel * 16);
}
__device__ __forceinline__ v8f wmma_bf16f32(v16bf a, v16bf b, v8f c) {
  return __builtin_amdgcn_wmma_f32_16x16x32_bf16(false, a, false, b, (short)0, c,
                                                 false, false);
}

// -------------------- f32 -> bf16 conversion --------------------
__global__ void __launch_bounds__(256) cvt_kernel(const float* __restrict__ s,
                                                  __bf16* __restrict__ d, int n) {
  int i = blockIdx.x * 256 + threadIdx.x;
  int st = gridDim.x * 256;
  for (; i < n; i += st) d[i] = (__bf16)s[i];
}

// Workgroup tile: 64(M) x 128(N), BK=32; 8 waves in 2x4 grid, each wave 32x32
// (4 accumulators, A/B fragment reuse -> 4 wmma per wave per K-step).

// ------------- Kernel 1: P = tanh(text_bf16 @ W_bf16^T), bf16 out -------------
__global__ void __launch_bounds__(256) proj_kernel(const __bf16* __restrict__ A,
                                                   const __bf16* __restrict__ Bt,
                                                   __bf16* __restrict__ P) {
  __shared__ __align__(16) __bf16 As[64][40];
  __shared__ __align__(16) __bf16 Bs[128][40];
  const int tid  = threadIdx.x;
  const int lane = tid & 31, w = tid >> 5;
  const int wm = w & 1, wn = w >> 1;
  const int lc = lane & 15, sel = lane >> 4;
  const int m0 = blockIdx.y * 64, n0 = blockIdx.x * 128;
  const int sm = tid >> 2, sk = (tid & 3) * 8;   // A staging
  const int sb = tid >> 1, sbk = (tid & 1) * 16; // B staging (two v8bf)

  v8f acc00 = {}, acc01 = {}, acc10 = {}, acc11 = {};
  const __bf16* arow0 = &As[wm * 32 + lc][0];
  const __bf16* arow1 = &As[wm * 32 + 16 + lc][0];
  const __bf16* bcol0 = &Bs[wn * 32 + lc][0];
  const __bf16* bcol1 = &Bs[wn * 32 + 16 + lc][0];

  for (int k0 = 0; k0 < H_; k0 += 32) {
    *(v8bf*)&As[sm][sk] = *(const v8bf*)(A + (size_t)(m0 + sm) * H_ + k0 + sk);
    const __bf16* bp = Bt + (size_t)(n0 + sb) * H_ + k0 + sbk;
    *(v8bf*)&Bs[sb][sbk]     = *(const v8bf*)(bp);
    *(v8bf*)&Bs[sb][sbk + 8] = *(const v8bf*)(bp + 8);
    if (k0 + 32 < H_) {
      __builtin_prefetch(A + (size_t)(m0 + sm) * H_ + k0 + 32 + sk, 0, 3);
      __builtin_prefetch(bp + 32, 0, 3);
    }
    __syncthreads();
    v16bf a0 = frag_a(arow0, sel);
    v16bf a1 = frag_a(arow1, sel);
    v16bf b0 = frag_b(bcol0, sel);
    v16bf b1 = frag_b(bcol1, sel);
    acc00 = wmma_bf16f32(a0, b0, acc00);
    acc01 = wmma_bf16f32(a0, b1, acc01);
    acc10 = wmma_bf16f32(a1, b0, acc10);
    acc11 = wmma_bf16f32(a1, b1, acc11);
    __syncthreads();
  }
  const int na = n0 + wn * 32 + lc, nb = na + 16;
  const int mb0 = m0 + wm * 32 + sel * 8, mb1 = mb0 + 16;
#pragma unroll
  for (int r = 0; r < 8; ++r) {
    P[(size_t)(mb0 + r) * H_ + na] = (__bf16)tanhf(acc00[r]);
    P[(size_t)(mb0 + r) * H_ + nb] = (__bf16)tanhf(acc01[r]);
    P[(size_t)(mb1 + r) * H_ + na] = (__bf16)tanhf(acc10[r]);
    P[(size_t)(mb1 + r) * H_ + nb] = (__bf16)tanhf(acc11[r]);
  }
}

// ------ Kernel 2: scores = P @ code^T, written TRANSPOSED (b,l,s) to d_out ------
__global__ void __launch_bounds__(256) scores_kernel(const __bf16* __restrict__ A,
                                                     const __bf16* __restrict__ Bt,
                                                     float* __restrict__ Out) {
  __shared__ __align__(16) __bf16 As[64][40];
  __shared__ __align__(16) __bf16 Bs[128][40];
  const int tid  = threadIdx.x;
  const int lane = tid & 31, w = tid >> 5;
  const int wm = w & 1, wn = w >> 1;
  const int lc = lane & 15, sel = lane >> 4;
  const int m0 = blockIdx.y * 64, n0 = blockIdx.x * 128;
  const int sm = tid >> 2, sk = (tid & 3) * 8;
  const int sb = tid >> 1, sbk = (tid & 1) * 16;
  const int ng = n0 + sb;

  v8f acc00 = {}, acc01 = {}, acc10 = {}, acc11 = {};
  const __bf16* arow0 = &As[wm * 32 + lc][0];
  const __bf16* arow1 = &As[wm * 32 + 16 + lc][0];
  const __bf16* bcol0 = &Bs[wn * 32 + lc][0];
  const __bf16* bcol1 = &Bs[wn * 32 + 16 + lc][0];

  for (int k0 = 0; k0 < H_; k0 += 32) {
    *(v8bf*)&As[sm][sk] = *(const v8bf*)(A + (size_t)(m0 + sm) * H_ + k0 + sk);
    v8bf bv0 = {}, bv1 = {};
    if (ng < L_) {
      const __bf16* bp = Bt + (size_t)ng * H_ + k0 + sbk;
      bv0 = *(const v8bf*)(bp);
      bv1 = *(const v8bf*)(bp + 8);
      if (k0 + 32 < H_) __builtin_prefetch(bp + 32, 0, 3);
    }
    *(v8bf*)&Bs[sb][sbk]     = bv0;
    *(v8bf*)&Bs[sb][sbk + 8] = bv1;
    if (k0 + 32 < H_)
      __builtin_prefetch(A + (size_t)(m0 + sm) * H_ + k0 + 32 + sk, 0, 3);
    __syncthreads();
    v16bf a0 = frag_a(arow0, sel);
    v16bf a1 = frag_a(arow1, sel);
    v16bf b0 = frag_b(bcol0, sel);
    v16bf b1 = frag_b(bcol1, sel);
    acc00 = wmma_bf16f32(a0, b0, acc00);
    acc01 = wmma_bf16f32(a0, b1, acc01);
    acc10 = wmma_bf16f32(a1, b0, acc10);
    acc11 = wmma_bf16f32(a1, b1, acc11);
    __syncthreads();
  }
  // lane holds fixed N (label) and 8 consecutive M (seq) -> contiguous (b,l,s) stores
  const int b    = m0 >> 12;
  const int sst0 = (m0 & (SEQ_ - 1)) + wm * 32 + sel * 8;
  const int sst1 = sst0 + 16;
  const int na   = n0 + wn * 32 + lc, nb = na + 16;
  if (na < L_) {
    float* q = Out + ((size_t)b * L_ + na) * SEQ_ + sst0;
    *(float4*)(q)      = make_float4(acc00[0], acc00[1], acc00[2], acc00[3]);
    *(float4*)(q + 4)  = make_float4(acc00[4], acc00[5], acc00[6], acc00[7]);
    *(float4*)(q + 16) = make_float4(acc10[0], acc10[1], acc10[2], acc10[3]);
    *(float4*)(q + 20) = make_float4(acc10[4], acc10[5], acc10[6], acc10[7]);
  }
  if (nb < L_) {
    float* q = Out + ((size_t)b * L_ + nb) * SEQ_ + sst0;
    *(float4*)(q)      = make_float4(acc01[0], acc01[1], acc01[2], acc01[3]);
    *(float4*)(q + 4)  = make_float4(acc01[4], acc01[5], acc01[6], acc01[7]);
    *(float4*)(q + 16) = make_float4(acc11[0], acc11[1], acc11[2], acc11[3]);
    *(float4*)(q + 20) = make_float4(acc11[4], acc11[5], acc11[6], acc11[7]);
  }
  (void)sst1;
}

// --------- Kernel 3: in-place softmax over the contiguous SEQ axis ---------
__global__ void __launch_bounds__(256) softmax_kernel(float* __restrict__ Att) {
  const int tid = threadIdx.x;
  float* p = Att + (size_t)blockIdx.x * SEQ_;
  float4 v[4];
  float mx = -3.402823466e+38f;
#pragma unroll
  for (int i = 0; i < 4; ++i) {
    v[i] = ((const float4*)p)[tid + i * 256];
    mx = fmaxf(mx, fmaxf(fmaxf(v[i].x, v[i].y), fmaxf(v[i].z, v[i].w)));
  }
  __shared__ float red[256];
  red[tid] = mx; __syncthreads();
  for (int s = 128; s > 0; s >>= 1) {
    if (tid < s) red[tid] = fmaxf(red[tid], red[tid + s]);
    __syncthreads();
  }
  mx = red[0]; __syncthreads();
  float sum = 0.f;
#pragma unroll
  for (int i = 0; i < 4; ++i) {
    v[i].x = __expf(v[i].x - mx); v[i].y = __expf(v[i].y - mx);
    v[i].z = __expf(v[i].z - mx); v[i].w = __expf(v[i].w - mx);
    sum += v[i].x + v[i].y + v[i].z + v[i].w;
  }
  red[tid] = sum; __syncthreads();
  for (int s = 128; s > 0; s >>= 1) {
    if (tid < s) red[tid] += red[tid + s];
    __syncthreads();
  }
  const float inv = 1.0f / red[0];
#pragma unroll
  for (int i = 0; i < 4; ++i) {
    v[i].x *= inv; v[i].y *= inv; v[i].z *= inv; v[i].w *= inv;
    ((float4*)p)[tid + i * 256] = v[i];
  }
}

// ----- Kernel 4: out[b,l,h] = att[b,l,:] @ text[b,:,h]; bf16 convert on stage -----
__global__ void __launch_bounds__(256) weighted_kernel(const float* __restrict__ Att,
                                                       const float* __restrict__ Text,
                                                       float* __restrict__ Out) {
  __shared__ __align__(16) __bf16 As[64][40];
  __shared__ __align__(16) __bf16 Bs[128][40];
  const int tid  = threadIdx.x;
  const int lane = tid & 31, w = tid >> 5;
  const int wm = w & 1, wn = w >> 1;
  const int lc = lane & 15, sel = lane >> 4;
  const int bz = blockIdx.z;
  const int m0 = blockIdx.y * 64;       // label tile
  const int n0 = blockIdx.x * 128;      // hidden tile
  const float* Ab = Att + (size_t)bz * L_ * SEQ_;
  const float* Tb = Text + (size_t)bz * SEQ_ * H_;
  const int sm = tid >> 2, sk = (tid & 3) * 8;     // A staging: row, k-chunk
  const int bn = tid & 127, bk = (tid >> 7) * 16;  // B staging: col(h), k-chunk(s)
  const int lg = m0 + sm;

  v8f acc00 = {}, acc01 = {}, acc10 = {}, acc11 = {};
  const __bf16* arow0 = &As[wm * 32 + lc][0];
  const __bf16* arow1 = &As[wm * 32 + 16 + lc][0];
  const __bf16* bcol0 = &Bs[wn * 32 + lc][0];
  const __bf16* bcol1 = &Bs[wn * 32 + 16 + lc][0];

  for (int k0 = 0; k0 < SEQ_; k0 += 32) {
    // A: att rows (f32 -> bf16)
    v8bf av = {};
    if (lg < L_) {
      const float* ap = Ab + (size_t)lg * SEQ_ + k0 + sk;
      float4 f0 = *(const float4*)ap;
      float4 f1 = *(const float4*)(ap + 4);
      av[0] = (__bf16)f0.x; av[1] = (__bf16)f0.y; av[2] = (__bf16)f0.z; av[3] = (__bf16)f0.w;
      av[4] = (__bf16)f1.x; av[5] = (__bf16)f1.y; av[6] = (__bf16)f1.z; av[7] = (__bf16)f1.w;
      if (k0 + 32 < SEQ_) __builtin_prefetch(ap + 32, 0, 3);
    }
    *(v8bf*)&As[sm][sk] = av;
    // B: transpose-stage text (k=s rows contiguous in h -> column-major LDS)
    v8bf bv0, bv1;
#pragma unroll
    for (int i = 0; i < 8; ++i) {
      bv0[i] = (__bf16)Tb[(size_t)(k0 + bk + i) * H_ + n0 + bn];
      bv1[i] = (__bf16)Tb[(size_t)(k0 + bk + 8 + i) * H_ + n0 + bn];
    }
    *(v8bf*)&Bs[bn][bk]     = bv0;
    *(v8bf*)&Bs[bn][bk + 8] = bv1;
    __syncthreads();
    v16bf a0 = frag_a(arow0, sel);
    v16bf a1 = frag_a(arow1, sel);
    v16bf b0 = frag_b(bcol0, sel);
    v16bf b1 = frag_b(bcol1, sel);
    acc00 = wmma_bf16f32(a0, b0, acc00);
    acc01 = wmma_bf16f32(a0, b1, acc01);
    acc10 = wmma_bf16f32(a1, b0, acc10);
    acc11 = wmma_bf16f32(a1, b1, acc11);
    __syncthreads();
  }
  const int na = n0 + wn * 32 + lc, nb = na + 16;
  const int lb0 = m0 + wm * 32 + sel * 8, lb1 = lb0 + 16;
#pragma unroll
  for (int r = 0; r < 8; ++r) {
    const int l0r = lb0 + r, l1r = lb1 + r;
    if (l0r < L_) {
      Out[((size_t)bz * L_ + l0r) * H_ + na] = acc00[r];
      Out[((size_t)bz * L_ + l0r) * H_ + nb] = acc01[r];
    }
    if (l1r < L_) {
      Out[((size_t)bz * L_ + l1r) * H_ + na] = acc10[r];
      Out[((size_t)bz * L_ + l1r) * H_ + nb] = acc11[r];
    }
  }
}

extern "C" void kernel_launch(void* const* d_in, const int* in_sizes, int n_in,
                              void* d_out, int out_size, void* d_ws, size_t ws_size,
                              hipStream_t stream) {
  const float* code = (const float*)d_in[0];   // (L, H)
  const float* text = (const float*)d_in[1];   // (BS, SEQ, H)
  // d_in[2] = attention_mask: all-ones, unused by the reference math
  const float* W    = (const float*)d_in[3];   // (H, H)

  float* out  = (float*)d_out;
  float* att  = out;                                   // (BS, L, SEQ)
  float* wout = out + (size_t)BS_ * L_ * SEQ_;         // (BS, L, H)

  __bf16* textb = (__bf16*)d_ws;                       // BS*SEQ*H
  __bf16* Wb    = textb + (size_t)BS_ * SEQ_ * H_;     // H*H
  __bf16* codeb = Wb + (size_t)H_ * H_;                // L*H
  __bf16* Pb    = codeb + (size_t)L_ * H_;             // BS*SEQ*H

  cvt_kernel<<<2048, 256, 0, stream>>>(text, textb, BS_ * SEQ_ * H_);
  cvt_kernel<<<512,  256, 0, stream>>>(W, Wb, H_ * H_);
  cvt_kernel<<<2048, 256, 0, stream>>>(code, codeb, L_ * H_);

  proj_kernel<<<dim3(H_ / 128, (BS_ * SEQ_) / 64), 256, 0, stream>>>(textb, Wb, Pb);
  scores_kernel<<<dim3((L_ + 127) / 128, (BS_ * SEQ_) / 64), 256, 0, stream>>>(Pb, codeb, att);
  softmax_kernel<<<BS_ * L_, 256, 0, stream>>>(att);
  weighted_kernel<<<dim3(H_ / 128, (L_ + 63) / 64, BS_), 256, 0, stream>>>(att, text, wout);
}
